// TGATlayer_26259430048438
// MI455X (gfx1250) — matrix operand order repaired and checked by reading, not compiled, not run on hardware
//
#include <hip/hip_runtime.h>
#include <hip/hip_fp16.h>

// CDNA5 / gfx1250 vector types for WMMA
typedef __attribute__((ext_vector_type(16))) _Float16 v16h;
typedef __attribute__((ext_vector_type(8)))  float    v8f;

#define NN 4096

// ---------------------------------------------------------------------------
// start conv: [1,N,24] -> [2,N,24]
__global__ void k_start(const float* __restrict__ nf, const float* __restrict__ w,
                        const float* __restrict__ b, float* __restrict__ x) {
  int idx = blockIdx.x * blockDim.x + threadIdx.x;   // n*24 + t
  if (idx >= NN * 24) return;
  float v = nf[idx];
  x[0 * NN * 24 + idx] = w[0] * v + b[0];
  x[1 * NN * 24 + idx] = w[1] * v + b[1];
}

// gated dilated conv: x[2][N][Win] -> y[2][N][Wout], kernel (1,2), dilation d
__global__ void k_gconv(const float* __restrict__ x, float* __restrict__ y,
                        const float* __restrict__ fw, const float* __restrict__ fb,
                        const float* __restrict__ gw, const float* __restrict__ gb,
                        int Win, int Wout, int d) {
  int idx = blockIdx.x * blockDim.x + threadIdx.x;
  if (idx >= NN * Wout) return;
  int n = idx / Wout, t = idx - n * Wout;
  float x00 = x[(0 * NN + n) * Win + t], x01 = x[(0 * NN + n) * Win + t + d];
  float x10 = x[(1 * NN + n) * Win + t], x11 = x[(1 * NN + n) * Win + t + d];
#pragma unroll
  for (int dc = 0; dc < 2; ++dc) {
    float f = fb[dc] + fw[dc*4+0]*x00 + fw[dc*4+1]*x01 + fw[dc*4+2]*x10 + fw[dc*4+3]*x11;
    float g = gb[dc] + gw[dc*4+0]*x00 + gw[dc*4+1]*x01 + gw[dc*4+2]*x10 + gw[dc*4+3]*x11;
    y[(dc * NN + n) * Wout + t] = tanhf(f) * (1.f / (1.f + __expf(-g)));
  }
}

// residual 1x1 conv path (layers 1,3): xpre = res_w*y + res_b + xin[...,-Wout:]
__global__ void k_resadd(const float* __restrict__ y, const float* __restrict__ xin,
                         float* __restrict__ xpre,
                         const float* __restrict__ rw, const float* __restrict__ rb,
                         int Win, int Wout) {
  int idx = blockIdx.x * blockDim.x + threadIdx.x;
  if (idx >= NN * Wout) return;
  int n = idx / Wout, t = idx - n * Wout;
  float y0 = y[(0 * NN + n) * Wout + t], y1 = y[(1 * NN + n) * Wout + t];
  int off = Win - Wout;
#pragma unroll
  for (int rc = 0; rc < 2; ++rc)
    xpre[(rc * NN + n) * Wout + t] =
        rb[rc] + rw[rc*2+0]*y0 + rw[rc*2+1]*y1 + xin[(rc * NN + n) * Win + off + t];
}

// BatchNorm training-mode stats (biased var) per channel
__global__ void k_bnstats(const float* __restrict__ xpre, int W, float* __restrict__ stats) {
  int c = blockIdx.x;
  const float* p = xpre + (size_t)c * NN * W;
  int tot = NN * W;
  float s = 0.f, s2 = 0.f;
  for (int i = threadIdx.x; i < tot; i += blockDim.x) { float v = p[i]; s += v; s2 += v * v; }
  __shared__ float sh[256], sh2[256];
  sh[threadIdx.x] = s; sh2[threadIdx.x] = s2; __syncthreads();
  for (int o = 128; o > 0; o >>= 1) {
    if ((int)threadIdx.x < o) { sh[threadIdx.x] += sh[threadIdx.x+o]; sh2[threadIdx.x] += sh2[threadIdx.x+o]; }
    __syncthreads();
  }
  if (threadIdx.x == 0) {
    float m = sh[0] / tot;
    stats[c * 2] = m;
    stats[c * 2 + 1] = sh2[0] / tot - m * m;
  }
}

__global__ void k_bnnorm(const float* __restrict__ xpre, float* __restrict__ xout,
                         const float* __restrict__ stats,
                         const float* __restrict__ g, const float* __restrict__ b, int W) {
  int idx = blockIdx.x * blockDim.x + threadIdx.x;
  if (idx >= 2 * NN * W) return;
  int c = idx / (NN * W);
  float m = stats[c * 2], is = rsqrtf(stats[c * 2 + 1] + 1e-5f);
  xout[idx] = (xpre[idx] - m) * is * g[c] + b[c];
}

// ---------------------------------------------------------------------------
// GAT stage 0: z = x@fc1^T, zi = x@fc2^T, score halves; zhT stored f16
// transposed+padded to 32 rows for WMMA B fragments.
template <int D>
__global__ void k_gat_prep(const float* __restrict__ y,
                           const float* __restrict__ fc1, const float* __restrict__ fc2,
                           const float* __restrict__ attn,
                           _Float16* __restrict__ zhT,   // [2][32][NN]
                           float* __restrict__ zi,       // [2][NN][32]
                           float* __restrict__ ssrc, float* __restrict__ sdst) {
  int n = blockIdx.x * blockDim.x + threadIdx.x;
  int c = blockIdx.y;
  if (n >= NN) return;
  const float* xr = y + ((size_t)c * NN + n) * D;
  float xl[D];
#pragma unroll
  for (int k = 0; k < D; ++k) xl[k] = xr[k];
  float ss = 0.f, sd = 0.f;
  _Float16* zt = zhT + (size_t)c * 32 * NN;
  float* zir = zi + ((size_t)c * NN + n) * 32;
  for (int dd = 0; dd < 32; ++dd) {
    float zv = 0.f;
    if (dd < D) {
      float ziv = 0.f;
      for (int k = 0; k < D; ++k) { zv += xl[k] * fc1[dd * D + k]; ziv += xl[k] * fc2[dd * D + k]; }
      ss += zv * attn[dd];
      sd += zv * attn[D + dd];
      zir[dd] = ziv;
    }
    zt[(size_t)dd * NN + n] = (_Float16)zv;   // rows >= D are zero padding
  }
  ssrc[c * NN + n] = ss;
  sdst[c * NN + n] = sd;
}

// GAT pass 1: per-row online max + sum(exp) over masked leaky-relu scores.
// First toucher of adjm (cold HBM) -> keep the prefetch stream here.
__global__ void k_gat_pass1(const float* __restrict__ adjm,
                            const float* __restrict__ ssrc, const float* __restrict__ sdst,
                            const float* __restrict__ fc0, const float* __restrict__ attn, int D,
                            float* __restrict__ mrow, float* __restrict__ lrow) {
  int lane = threadIdx.x;
  int row = blockIdx.x * blockDim.y + threadIdx.y;
  int c = blockIdx.y;
  float cconst = fc0[0] * attn[2 * D];
  float sdr = sdst[c * NN + row];
  const float* ar = adjm + (size_t)row * NN;
  const float* sr = ssrc + c * NN;
  float m = -1e30f, l = 0.f;
  for (int j = lane; j < NN; j += 32) {
    float a = ar[j];
    __builtin_prefetch(ar + j + 256, 0, 1);   // gfx1250 global_prefetch_b8
    if (a != 0.f) {                            // ~1.6% density: execz-skip wins
      float e = sdr + sr[j] + a * cconst;
      e = e > 0.f ? e : 0.01f * e;
      if (e > m) { l = l * __expf(m - e) + 1.f; m = e; }
      else       { l += __expf(e - m); }
    }
  }
#pragma unroll
  for (int o = 16; o > 0; o >>= 1) {
    float om = __shfl_xor(m, o, 32);
    float ol = __shfl_xor(l, o, 32);
    float nm = fmaxf(m, om);
    l = l * __expf(m - nm) + ol * __expf(om - nm);
    m = nm;
  }
  if (lane == 0) { mrow[c * NN + row] = m; lrow[c * NN + row] = l; }
}

// GAT pass 2: flash-style aggregation via v_wmma_f32_16x16x32_f16.
// One wave owns a 16-row tile; A = alpha (f16, transposed through a PRIVATE
// per-wave LDS slice), B = z^T (f16, straight from L2).
// No block barriers: LDS is per-wave and the DS pipeline is in-order per wave,
// so __builtin_amdgcn_wave_barrier() (compiler-only fence) is sufficient to
// order the cross-lane store->load; the 8 waves free-run on independent tiles.
template <int D>
__global__ __launch_bounds__(256) void k_gat_pass2(
    const float* __restrict__ adjm,
    const float* __restrict__ ssrc, const float* __restrict__ sdst,
    const float* __restrict__ fc0, const float* __restrict__ attn,
    const float* __restrict__ mrow, const float* __restrict__ lrow,
    const _Float16* __restrict__ zhT, const float* __restrict__ zi,
    const float* __restrict__ xin, int Win, float* __restrict__ xpre) {
  constexpr bool TWO = (D > 16);
  __shared__ _Float16 aT[8][16][32];
  int lane = threadIdx.x & 31;
  int w = threadIdx.x >> 5;
  int c = blockIdx.y;
  int i0 = (blockIdx.x * 8 + w) * 16;
  const float* sr = ssrc + c * NN;
  const float* sd = sdst + c * NN;
  const float* mr = mrow + c * NN;
  const _Float16* zt = zhT + (size_t)c * 32 * NN;
  _Float16* atw = &aT[w][0][0];
  float cconst = fc0[0] * attn[2 * D];
  float sdl[16], ml[16];
#pragma unroll
  for (int rr = 0; rr < 16; ++rr) { sdl[rr] = sd[i0 + rr]; ml[rr] = mr[i0 + rr]; }

  v8f acc0 = {}; v8f acc1 = {};
  int m16 = lane & 15;
  int hi = lane >> 4;

  for (int jc = 0; jc < NN; jc += 32) {
    float sj = sr[jc + lane];
#pragma unroll
    for (int rr = 0; rr < 16; ++rr) {          // coalesced adjm row reads, lane = column
      float a = adjm[(size_t)(i0 + rr) * NN + jc + lane];
      float av = 0.f;
      if (a != 0.f) {                          // sparse: execz-skip most chunks
        float e = sdl[rr] + sj + a * cconst;
        e = e > 0.f ? e : 0.01f * e;
        av = __expf(e - ml[rr]);               // unnormalized alpha in [0,1]
      }
      atw[rr * 32 + lane] = (_Float16)av;
    }
    __builtin_amdgcn_wave_barrier();           // order stores before fragment loads

    // A fragment: K contiguous (lane<16: K 0-15; lane>=16: K 16-31), matching B
    v16h af;
    const _Float16* arow = atw + m16 * 32;
    int b0 = hi ? 16 : 0;
#pragma unroll
    for (int t = 0; t < 16; ++t) af[t] = arow[b0 + t];
    __builtin_amdgcn_wave_barrier();           // keep next iter's stores after these loads

    // B fragment: lane n holds column n, K contiguous (0-15 / 16-31) from z^T
    int k0 = hi ? 16 : 0;
    const _Float16* bp0 = zt + (size_t)m16 * NN + jc + k0;
    v16h bf0;
#pragma unroll
    for (int t = 0; t < 16; ++t) bf0[t] = bp0[t];
    acc0 = __builtin_amdgcn_wmma_f32_16x16x32_f16(false, af, false, bf0,
                                                  (short)0, acc0, false, false);
    if constexpr (TWO) {
      const _Float16* bp1 = zt + (size_t)(16 + m16) * NN + jc + k0;
      v16h bf1;
#pragma unroll
      for (int t = 0; t < 16; ++t) bf1[t] = bp1[t];
      acc1 = __builtin_amdgcn_wmma_f32_16x16x32_f16(false, af, false, bf1,
                                                    (short)0, acc1, false, false);
    }
  }

  // Epilogue per C/D layout: VGPR r -> M = r + 8*hi, N = lane&15
  int toff = Win - D;
#pragma unroll
  for (int r = 0; r < 8; ++r) {
    int M = r + hi * 8;
    int row = i0 + M;
    float invl = 1.f / lrow[c * NN + row];
    if (m16 < D) {
      float h0 = fmaxf(acc0[r] * invl + zi[((size_t)c * NN + row) * 32 + m16], 0.f);
      xpre[((size_t)c * NN + row) * D + m16] =
          h0 + xin[((size_t)c * NN + row) * Win + toff + m16];
    }
    if constexpr (TWO) {
      int col1 = 16 + m16;
      if (col1 < D) {
        float h1 = fmaxf(acc1[r] * invl + zi[((size_t)c * NN + row) * 32 + col1], 0.f);
        xpre[((size_t)c * NN + row) * D + col1] =
            h1 + xin[((size_t)c * NN + row) * Win + toff + col1];
      }
    }
  }
}

// final: relu -> end1 (2->20) -> relu -> end2 (20->24), width 1
__global__ void k_final(const float* __restrict__ x,
                        const float* __restrict__ e1w, const float* __restrict__ e1b,
                        const float* __restrict__ e2w, const float* __restrict__ e2b,
                        float* __restrict__ out) {
  int n = blockIdx.x * blockDim.x + threadIdx.x;
  if (n >= NN) return;
  float x0 = fmaxf(x[n], 0.f), x1 = fmaxf(x[NN + n], 0.f);
  float h[20];
#pragma unroll
  for (int o = 0; o < 20; ++o) h[o] = fmaxf(e1b[o] + e1w[o*2]*x0 + e1w[o*2+1]*x1, 0.f);
#pragma unroll
  for (int o = 0; o < 24; ++o) {
    float s = e2b[o];
    for (int k = 0; k < 20; ++k) s += e2w[o * 20 + k] * h[k];
    out[(size_t)o * NN + n] = s;
  }
}

// ---------------------------------------------------------------------------
template <int D>
static void launch_gat(const float* adjm, const float* y,
                       const float* fc1, const float* fc2,
                       const float* fc0, const float* attn,
                       _Float16* zhT, float* zi, float* ssrc, float* sdst,
                       float* mrow, float* lrow,
                       const float* xin, int Win, float* xpre, hipStream_t stream) {
  k_gat_prep<D><<<dim3(NN / 128, 2), 128, 0, stream>>>(y, fc1, fc2, attn, zhT, zi, ssrc, sdst);
  k_gat_pass1<<<dim3(NN / 8, 2), dim3(32, 8), 0, stream>>>(adjm, ssrc, sdst, fc0, attn, D,
                                                           mrow, lrow);
  k_gat_pass2<D><<<dim3(NN / 128, 2), 256, 0, stream>>>(adjm, ssrc, sdst, fc0, attn,
                                                        mrow, lrow, zhT, zi, xin, Win, xpre);
}

extern "C" void kernel_launch(void* const* d_in, const int* in_sizes, int n_in,
                              void* d_out, int out_size, void* d_ws, size_t ws_size,
                              hipStream_t stream) {
  (void)in_sizes; (void)n_in; (void)out_size; (void)ws_size;
  const float* adjm = (const float*)d_in[0];
  const float* nf   = (const float*)d_in[1];
  auto P = [&](int i) { return (const float*)d_in[i]; };
  // params flattened jax-tree style: sorted keys, lists in order
  // 2..6 bn_b[0..4] | 7..11 bn_g | 12 end1_b | 13 end1_w | 14 end2_b | 15 end2_w
  // 16..20 filt_b | 21..25 filt_w | 26..37 gat{attn,fc0,fc1,fc2}x3
  // 38..42 gate_b | 43..47 gate_w | 48..52 res_b | 53..57 res_w | 58 start_b | 59 start_w
  const float *bn_b[5], *bn_g[5], *filt_b[5], *filt_w[5], *gate_b[5], *gate_w[5],
      *res_b[5], *res_w[5];
  for (int i = 0; i < 5; ++i) {
    bn_b[i] = P(2 + i);  bn_g[i] = P(7 + i);
    filt_b[i] = P(16 + i); filt_w[i] = P(21 + i);
    gate_b[i] = P(38 + i); gate_w[i] = P(43 + i);
    res_b[i] = P(48 + i);  res_w[i] = P(53 + i);
  }
  const float *end1_b = P(12), *end1_w = P(13), *end2_b = P(14), *end2_w = P(15);
  const float *g_attn[3] = {P(26), P(30), P(34)};
  const float *g_fc0[3]  = {P(27), P(31), P(35)};
  const float *g_fc1[3]  = {P(28), P(32), P(36)};
  const float *g_fc2[3]  = {P(29), P(33), P(37)};
  const float *start_b = P(58), *start_w = P(59);

  char* wsb = (char*)d_ws;
  auto alloc = [&](size_t bytes) {
    char* p = wsb; wsb += (bytes + 255) & ~(size_t)255; return p;
  };
  float*    xA    = (float*)alloc(2 * NN * 24 * sizeof(float));
  float*    xB    = (float*)alloc(2 * NN * 24 * sizeof(float));
  float*    ybuf  = (float*)alloc(2 * NN * 24 * sizeof(float));
  float*    xpre  = (float*)alloc(2 * NN * 24 * sizeof(float));
  _Float16* zhT   = (_Float16*)alloc(2 * 32 * NN * sizeof(_Float16));
  float*    zi    = (float*)alloc(2 * (size_t)NN * 32 * sizeof(float));
  float*    ssrc  = (float*)alloc(2 * NN * sizeof(float));
  float*    sdst  = (float*)alloc(2 * NN * sizeof(float));
  float*    mrow  = (float*)alloc(2 * NN * sizeof(float));
  float*    lrow  = (float*)alloc(2 * NN * sizeof(float));
  float*    stats = (float*)alloc(4 * sizeof(float));

  const int dils[5] = {1, 2, 4, 8, 8};
  float* cur = xA;
  float* nxt = xB;
  k_start<<<(NN * 24 + 255) / 256, 256, 0, stream>>>(nf, start_w, start_b, cur);

  int Win = 24, gi = 0;
  for (int i = 0; i < 5; ++i) {
    int d = dils[i], Wout = Win - d;
    k_gconv<<<(NN * Wout + 255) / 256, 256, 0, stream>>>(
        cur, ybuf, filt_w[i], filt_b[i], gate_w[i], gate_b[i], Win, Wout, d);
    if ((i & 1) == 0) {
      if (gi == 0)
        launch_gat<23>(adjm, ybuf, g_fc1[0], g_fc2[0], g_fc0[0], g_attn[0],
                       zhT, zi, ssrc, sdst, mrow, lrow, cur, Win, xpre, stream);
      else if (gi == 1)
        launch_gat<17>(adjm, ybuf, g_fc1[1], g_fc2[1], g_fc0[1], g_attn[1],
                       zhT, zi, ssrc, sdst, mrow, lrow, cur, Win, xpre, stream);
      else
        launch_gat<1>(adjm, ybuf, g_fc1[2], g_fc2[2], g_fc0[2], g_attn[2],
                      zhT, zi, ssrc, sdst, mrow, lrow, cur, Win, xpre, stream);
      ++gi;
    } else {
      k_resadd<<<(NN * Wout + 255) / 256, 256, 0, stream>>>(
          ybuf, cur, xpre, res_w[i], res_b[i], Win, Wout);
    }
    k_bnstats<<<2, 256, 0, stream>>>(xpre, Wout, stats);
    k_bnnorm<<<(2 * NN * Wout + 255) / 256, 256, 0, stream>>>(
        xpre, nxt, stats, bn_g[i], bn_b[i], Wout);
    float* t = cur; cur = nxt; nxt = t;
    Win = Wout;
  }
  k_final<<<(NN + 255) / 256, 256, 0, stream>>>(cur, end1_w, end1_b, end2_w, end2_b,
                                                (float*)d_out);
}
